// LinearAttention_670014899003
// MI455X (gfx1250) — matrix operand order repaired and checked by reading, not compiled, not run on hardware
//
#include <hip/hip_runtime.h>
#include <hip/hip_bf16.h>
#include <stddef.h>

// Problem constants (from reference)
#define B_     8
#define C_     256      // DIM
#define N_     4096     // H*W
#define HID_   512
#define OQKV_  1536     // 3*HID
#define HEADS_ 8
#define DH_    64

typedef __attribute__((ext_vector_type(16))) __bf16        bf16x16;
typedef __attribute__((ext_vector_type(8)))  float         f32x8;
typedef __attribute__((ext_vector_type(4)))  unsigned int  u32x4;
typedef __attribute__((ext_vector_type(8)))  unsigned int  u32x8;

__device__ __forceinline__ unsigned short f2bf(float f) {
  unsigned u = __float_as_uint(f);
  u += 0x7FFFu + ((u >> 16) & 1u);           // round-to-nearest-even
  return (unsigned short)(u >> 16);
}
__device__ __forceinline__ float bf2f(unsigned short s) {
  return __uint_as_float((unsigned)s << 16);
}
__device__ __forceinline__ float elu1(float x) {
  return x > 0.0f ? x + 1.0f : __expf(x);
}
// Assemble a 16-element bf16 WMMA fragment from two 16-byte chunks.
__device__ __forceinline__ bf16x16 load_frag(const unsigned short* p0,
                                             const unsigned short* p1) {
  u32x4 lo = *(const u32x4*)p0;
  u32x4 hi = *(const u32x4*)p1;
  u32x8 r;
  r[0]=lo[0]; r[1]=lo[1]; r[2]=lo[2]; r[3]=lo[3];
  r[4]=hi[0]; r[5]=hi[1]; r[6]=hi[2]; r[7]=hi[3];
  return __builtin_bit_cast(bf16x16, r);
}

// ---------------------------------------------------------------------------
// Kernel 0: bf16 weight conversion + zero the atomic accumulators (per launch)
// ---------------------------------------------------------------------------
__global__ void la_prep(const float* __restrict__ Wqkv, const float* __restrict__ Wout,
                        unsigned short* __restrict__ wqkv_bf,
                        unsigned short* __restrict__ wout_bf,
                        float* __restrict__ kv, float* __restrict__ ksum) {
  int idx = blockIdx.x * 256 + threadIdx.x;
  if (idx < OQKV_ * C_) {
    wqkv_bf[idx] = f2bf(Wqkv[idx]);
  } else {
    int j = idx - OQKV_ * C_;
    if (j < C_ * HID_) wout_bf[j] = f2bf(Wout[j]);
  }
  if (idx < B_ * HEADS_ * DH_ * DH_) kv[idx] = 0.0f;
  if (idx < B_ * HEADS_ * DH_)       ksum[idx] = 0.0f;
}

// ---------------------------------------------------------------------------
// Kernel 1: qkv = Wqkv @ x  (bf16 WMMA, f32 accum) + fused elu / contact-scale
// Block = 256 threads (8 waves), tile = 64(o) x 64(n), K loop over C in 32s.
// ---------------------------------------------------------------------------
#define XT_S 40   // LDS x-tile row stride in bf16 (32 + 8 pad -> 80B, 16B-mult)
__global__ void la_qkv(const float* __restrict__ x, const float* __restrict__ cm,
                       const unsigned short* __restrict__ wqkv_bf,
                       const float* __restrict__ csp,
                       unsigned short* __restrict__ qg, unsigned short* __restrict__ kg,
                       unsigned short* __restrict__ vg) {
  extern __shared__ __align__(16) char smem[];
  unsigned short* xt = (unsigned short*)smem;     // [64 n][XT_S k] bf16
  const int t = threadIdx.x;
  const int nb = blockIdx.x * 64;
  const int ob = blockIdx.y * 64;
  const int b  = blockIdx.z;
  const int wave = t >> 5, lane = t & 31;
  const int lrow = lane & 15, lhi = lane >> 4;
  const int o_sub = wave & 3, n_half = wave >> 2;

  f32x8 acc0 = {}, acc1 = {};
  const unsigned short* arow =
      wqkv_bf + (size_t)(ob + o_sub * 16 + lrow) * C_ + lhi * 8;

  for (int kk = 0; kk < C_; kk += 32) {
    __syncthreads();
    // stage x tile transposed: global (c-major) -> LDS (n, k) bf16
    #pragma unroll
    for (int i = 0; i < 8; ++i) {
      int cl = (t >> 6) + i * 4;     // 0..31
      int nl = t & 63;
      float xv = x[((size_t)b * C_ + kk + cl) * N_ + nb + nl];
      xt[nl * XT_S + cl] = f2bf(xv);
    }
    __syncthreads();

    bf16x16 A = load_frag(arow + kk, arow + kk + 16);
    const unsigned short* b0p = xt + ((n_half * 2 + 0) * 16 + lrow) * XT_S + lhi * 8;
    const unsigned short* b1p = xt + ((n_half * 2 + 1) * 16 + lrow) * XT_S + lhi * 8;
    bf16x16 B0 = load_frag(b0p, b0p + 16);
    bf16x16 B1 = load_frag(b1p, b1p + 16);
    acc0 = __builtin_amdgcn_wmma_f32_16x16x32_bf16(false, A, false, B0, (short)0, acc0, false, false);
    acc1 = __builtin_amdgcn_wmma_f32_16x16x32_bf16(false, A, false, B1, (short)0, acc1, false, false);
  }

  // fused epilogue: section 0 = q (elu), 1 = k (elu * contact), 2 = v (copy)
  const int sec = ob >> 9;
  const int orow0 = (ob & 511) + o_sub * 16 + 8 * lhi;
  unsigned short* dst = (sec == 0) ? qg : (sec == 1 ? kg : vg);
  const float cs = csp[0];
  #pragma unroll
  for (int j = 0; j < 2; ++j) {
    f32x8 a = j ? acc1 : acc0;
    int n = nb + (n_half * 2 + j) * 16 + lrow;
    float factor = 1.0f;
    if (sec == 1) factor = 1.0f + cs * cm[(size_t)b * N_ + n];
    #pragma unroll
    for (int v = 0; v < 8; ++v) {
      float val = a[v];
      if (sec == 0)      val = elu1(val);
      else if (sec == 1) val = elu1(val) * factor;
      dst[((size_t)b * HID_ + orow0 + v) * N_ + n] = f2bf(val);
    }
  }
}

// ---------------------------------------------------------------------------
// Kernel 2: kv[b,h] = k vT (64x64) and ksum[b,h] (memory-bound, fp32 VALU)
// grid (8 n-chunks, 64 b*h); atomic reduction into ws (zeroed by la_prep).
// ---------------------------------------------------------------------------
__global__ void la_kv(const unsigned short* __restrict__ kg,
                      const unsigned short* __restrict__ vg,
                      float* __restrict__ kv, float* __restrict__ ksum) {
  extern __shared__ __align__(16) char smem[];
  float* kld = (float*)smem;        // [64 d][65]
  float* vld = kld + 64 * 65;       // [64 e][65]
  const int t = threadIdx.x;
  const int bh = blockIdx.y;
  const int n0 = blockIdx.x * 512;
  const int e = t & 63, dg = t >> 6;
  float acc[16];
  #pragma unroll
  for (int j = 0; j < 16; ++j) acc[j] = 0.0f;
  float kssum = 0.0f;
  const int b = bh >> 3, h = bh & 7;
  const size_t rowbase = ((size_t)b * HID_ + h * DH_) * N_;

  for (int c = 0; c < 512; c += 64) {
    __syncthreads();
    #pragma unroll
    for (int i = 0; i < 16; ++i) {
      int nl = t & 63, r = (t >> 6) + i * 4;
      size_t g = rowbase + (size_t)r * N_ + n0 + c + nl;
      kld[r * 65 + nl] = bf2f(kg[g]);
      vld[r * 65 + nl] = bf2f(vg[g]);
    }
    __syncthreads();
    for (int nl = 0; nl < 64; ++nl) {
      float vv = vld[e * 65 + nl];
      #pragma unroll
      for (int j = 0; j < 16; ++j) acc[j] += kld[(dg * 16 + j) * 65 + nl] * vv;
    }
    if (t < 64) {
      float s = 0.0f;
      for (int nl = 0; nl < 64; ++nl) s += kld[t * 65 + nl];
      kssum += s;
    }
  }
  float* kvb = kv + (size_t)bh * DH_ * DH_;
  #pragma unroll
  for (int j = 0; j < 16; ++j) atomicAdd(&kvb[(dg * 16 + j) * DH_ + e], acc[j]);
  if (t < 64) atomicAdd(&ksum[(size_t)bh * DH_ + t], kssum);
}

// ---------------------------------------------------------------------------
// Kernel 3: fused  out = (kvT q) * z  per head  ->  y = Wout @ out + bout
// Block = 256 threads, tile_n = 32. Two WMMA stages through a padded LDS tile.
// ---------------------------------------------------------------------------
#define KT_S 72    // kvT/qT row stride (64 + 8 pad) -> 144B, 16B-mult
#define OT_S 520   // outT row stride (512 + 8 pad)  -> 1040B, 16B-mult
__global__ void la_out(const unsigned short* __restrict__ qg,
                       const unsigned short* __restrict__ wout_bf,
                       const float* __restrict__ kv, const float* __restrict__ ksum,
                       const float* __restrict__ bout, float* __restrict__ y) {
  extern __shared__ __align__(16) char smem[];
  unsigned short* kvt  = (unsigned short*)smem;          // [64 e][KT_S d]
  unsigned short* qt   = kvt + 64 * KT_S;                // [32 n][KT_S d]
  float*          zbuf = (float*)(qt + 32 * KT_S);       // [32]
  unsigned short* outt = (unsigned short*)(zbuf + 32);   // [32 n][OT_S hid]
  const int t = threadIdx.x;
  const int nb = blockIdx.x * 32;
  const int b  = blockIdx.y;
  const int wave = t >> 5, lane = t & 31;
  const int lrow = lane & 15, lhi = lane >> 4;
  const int e_sub = wave & 3, n_sub = wave >> 2;   // stage-A mapping (4x2 subtiles)

  for (int h = 0; h < HEADS_; ++h) {
    __syncthreads();
    // stage kvT (e,d) bf16 from global kv (d,e) f32
    const float* kvh = kv + (size_t)(b * HEADS_ + h) * DH_ * DH_;
    #pragma unroll
    for (int i = 0; i < 16; ++i) {
      int el = t & 63, d = (t >> 6) + i * 4;
      kvt[el * KT_S + d] = f2bf(kvh[d * DH_ + el]);
    }
    // stage qT (n,d) bf16 from global q (d,n) bf16
    const unsigned short* qh = qg + ((size_t)b * HID_ + h * DH_) * N_ + nb;
    #pragma unroll
    for (int i = 0; i < 8; ++i) {
      int nl = t & 31, d = (t >> 5) + i * 8;
      qt[nl * KT_S + d] = qh[(size_t)d * N_ + nl];
    }
    __syncthreads();
    // z[n] = 1 / (q . ksum + 1e-6)
    if (t < 32) {
      const float* ksh = ksum + (size_t)(b * HEADS_ + h) * DH_;
      float s = 0.0f;
      for (int d = 0; d < DH_; ++d) s += bf2f(qt[t * KT_S + d]) * ksh[d];
      zbuf[t] = 1.0f / (s + 1e-6f);
    }
    __syncthreads();
    // stage A WMMA: out_h[e][n] = sum_d kvT[e][d] * qT[n][d]
    f32x8 acc = {};
    #pragma unroll
    for (int kk = 0; kk < DH_; kk += 32) {
      const unsigned short* ap = kvt + (e_sub * 16 + lrow) * KT_S + kk + lhi * 8;
      const unsigned short* bp = qt  + (n_sub * 16 + lrow) * KT_S + kk + lhi * 8;
      bf16x16 A  = load_frag(ap, ap + 16);
      bf16x16 Bf = load_frag(bp, bp + 16);
      acc = __builtin_amdgcn_wmma_f32_16x16x32_bf16(false, A, false, Bf, (short)0, acc, false, false);
    }
    // scale by z and deposit into outT (n, hid) as packed bf16 pairs
    int nl = n_sub * 16 + lrow;
    float zv = zbuf[nl];
    unsigned* orow = (unsigned*)(outt + nl * OT_S + h * DH_ + e_sub * 16 + 8 * lhi);
    #pragma unroll
    for (int p = 0; p < 4; ++p) {
      unsigned u = (unsigned)f2bf(acc[2 * p] * zv) |
                   ((unsigned)f2bf(acc[2 * p + 1] * zv) << 16);
      orow[p] = u;
    }
  }
  __syncthreads();

  // stage B WMMA: y[c][n] = sum_hid Wout[c][hid] * outT[n][hid]
  const int n_sub2 = wave & 1;
  const int cgrp   = wave >> 1;   // 4 c-subtiles each
  f32x8 acc2[4] = {{}, {}, {}, {}};
  for (int kk = 0; kk < HID_; kk += 32) {
    const unsigned short* bp = outt + (n_sub2 * 16 + lrow) * OT_S + kk + lhi * 8;
    bf16x16 Bf = load_frag(bp, bp + 16);
    #pragma unroll
    for (int ci = 0; ci < 4; ++ci) {
      const unsigned short* ap =
          wout_bf + (size_t)((cgrp * 4 + ci) * 16 + lrow) * HID_ + kk + lhi * 8;
      bf16x16 A = load_frag(ap, ap + 16);
      acc2[ci] = __builtin_amdgcn_wmma_f32_16x16x32_bf16(false, A, false, Bf, (short)0, acc2[ci], false, false);
    }
  }
  int n = nb + n_sub2 * 16 + lrow;
  #pragma unroll
  for (int ci = 0; ci < 4; ++ci) {
    #pragma unroll
    for (int v = 0; v < 8; ++v) {
      int c = (cgrp * 4 + ci) * 16 + v + 8 * lhi;
      y[((size_t)b * C_ + c) * N_ + n] = acc2[ci][v] + bout[c];
    }
  }
}

// ---------------------------------------------------------------------------
// Host launch. ws layout (bytes, 256-aligned):
//   0         Wqkv bf16   786432
//   786432    Wout bf16   262144
//   1048576   q bf16      33554432
//   ...       k bf16      33554432
//   ...       v bf16      33554432
//   ...       kv f32      1048576
//   ...       ksum f32    16384     -> total ~102.8 MB
// ---------------------------------------------------------------------------
extern "C" void kernel_launch(void* const* d_in, const int* in_sizes, int n_in,
                              void* d_out, int out_size, void* d_ws, size_t ws_size,
                              hipStream_t stream) {
  (void)in_sizes; (void)n_in; (void)out_size; (void)ws_size;
  const float* x    = (const float*)d_in[0];
  const float* cm   = (const float*)d_in[1];
  const float* Wqkv = (const float*)d_in[2];
  const float* Wout = (const float*)d_in[3];
  const float* bout = (const float*)d_in[4];
  const float* csp  = (const float*)d_in[5];
  float* y = (float*)d_out;

  char* ws = (char*)d_ws;
  unsigned short* wqkv_bf = (unsigned short*)(ws);
  unsigned short* wout_bf = (unsigned short*)(ws + 786432);
  unsigned short* qg = (unsigned short*)(ws + 1048576);
  unsigned short* kg = (unsigned short*)(ws + 1048576 + 33554432ull);
  unsigned short* vg = (unsigned short*)(ws + 1048576 + 2ull * 33554432ull);
  float* kvb = (float*)(ws + 1048576 + 3ull * 33554432ull);
  float* ksb = (float*)(ws + 1048576 + 3ull * 33554432ull + 1048576ull);

  la_prep<<<2048, 256, 0, stream>>>(Wqkv, Wout, wqkv_bf, wout_bf, kvb, ksb);
  la_qkv<<<dim3(64, 24, 8), 256, 64 * XT_S * 2, stream>>>(x, cm, wqkv_bf, csp, qg, kg, vg);
  la_kv<<<dim3(8, 64), 256, 2 * 64 * 65 * 4, stream>>>(kg, vg, kvb, ksb);
  la_out<<<dim3(128, 8), 256, (64 * KT_S + 32 * KT_S) * 2 + 32 * 4 + 32 * OT_S * 2, stream>>>(
      qg, wout_bf, kvb, ksb, bout, y);
}
